// MultiHeadAttention_68719476736650
// MI455X (gfx1250) — compile-verified, matrix operand order checked by reading, not run on hardware
//
#include <hip/hip_runtime.h>
#include <hip/hip_bf16.h>
#include <math.h>

// MI455X / gfx1250: wave32, WMMA 16x16x32 bf16 -> f32 accumulate.
typedef __bf16 bf16_t;
typedef __attribute__((ext_vector_type(16))) __bf16 v16bf;
typedef __attribute__((ext_vector_type(8)))  __bf16 v8bf;
typedef __attribute__((ext_vector_type(4)))  __bf16 v4bf;
typedef __attribute__((ext_vector_type(8)))  float  v8f;
typedef __attribute__((ext_vector_type(4)))  float  v4f;
typedef __attribute__((ext_vector_type(4)))  int    v4i;

#define BB  2
#define SS  2048
#define DD  1024
#define HH  16
#define DKK 64
#define MM  (BB*SS)   // 4096 rows

__device__ __forceinline__ v8f zero_v8f() {
  v8f z;
#pragma unroll
  for (int i = 0; i < 8; ++i) z[i] = 0.0f;
  return z;
}

__device__ __forceinline__ v8f wmma_bf16(v16bf a, v16bf b, v8f c) {
  // (neg_a, A, neg_b, B, c_mod, C, reuse_a, reuse_b)
  return __builtin_amdgcn_wmma_f32_16x16x32_bf16(false, a, false, b,
                                                 (short)0, c, false, false);
}

// Depth-4 tree reductions (keep FP reassociation explicit in source).
__device__ __forceinline__ float vmax8(v8f a) {
  const float m0 = fmaxf(a[0], a[1]), m1 = fmaxf(a[2], a[3]);
  const float m2 = fmaxf(a[4], a[5]), m3 = fmaxf(a[6], a[7]);
  return fmaxf(fmaxf(m0, m1), fmaxf(m2, m3));
}

// A/B fragment per-lane layout (16-bit, 16x32 resp. 32x16):
//   lane L: half = L/16, row/col = L%16
//   frag elems [0..7]  = K = half*8 + 0..7
//   frag elems [8..15] = K = 16 + half*8 + 0..7
// Caller passes p = row_base + half*8; groups at p[0..7] and p[16..23].
__device__ __forceinline__ v16bf frag_from_bf16(const bf16_t* __restrict__ p) {
  v8bf lo = *(const v8bf*)(p);
  v8bf hi = *(const v8bf*)(p + 16);
  v16bf r;
#pragma unroll
  for (int i = 0; i < 8; ++i) { r[i] = lo[i]; r[i + 8] = hi[i]; }
  return r;
}

// Bulk f32 -> bf16 conversion, 4 elements per thread.
__global__ __launch_bounds__(256)
void cvt_bf16(const float* __restrict__ in, bf16_t* __restrict__ outp, int n4) {
  const int i = blockIdx.x * 256 + threadIdx.x;
  if (i < n4) {
    v4f a = ((const v4f*)in)[i];
    v4bf b;
#pragma unroll
    for (int j = 0; j < 4; ++j) b[j] = (__bf16)a[j];
    ((v4bf*)outp)[i] = b;
  }
}

// int mask -> additive f32 bias (0 or -1e9), 4 elements per thread.
__global__ __launch_bounds__(256)
void mask_bias(const int* __restrict__ mask, float* __restrict__ outp, int n4) {
  const int i = blockIdx.x * 256 + threadIdx.x;
  if (i < n4) {
    v4i m = ((const v4i*)mask)[i];
    v4f b;
#pragma unroll
    for (int j = 0; j < 4; ++j) b[j] = (m[j] == 0) ? -1.0e9f : 0.0f;
    ((v4f*)outp)[i] = b;
  }
}

// out = x @ W^T + bias ; x:[MM,DD] bf16, W:[DD,DD] bf16 (row n = out feature).
// Each wave: 32(M) x 128(N) tile, K stepped by 32 -> 16 WMMA per step,
// 20 b128 loads per step (W stays L2-resident).
// MODE 0: store bf16 head-split Qh/Kh [b,h,s,dk]
// MODE 1: store bf16 transposed   Vt [b,h,dk,s]
// MODE 2: store f32 row-major [MM,DD] (final output)
template <int MODE>
__global__ __launch_bounds__(256)
void proj_gemm(const bf16_t* __restrict__ x, const bf16_t* __restrict__ W,
               const float* __restrict__ bias,
               bf16_t* __restrict__ out_bf, float* __restrict__ out_f) {
  const int lane = threadIdx.x & 31;
  const int wave = threadIdx.x >> 5;
  const int tile = blockIdx.x * 8 + wave;     // (MM/32)*(DD/128) = 1024 tiles
  const int tn = tile & 7;                    // DD/128 = 8 n-tiles
  const int tm = tile >> 3;                   // 128 m-tiles
  const int m0 = tm * 32, n0 = tn * 128;
  const int half = lane >> 4, lm = lane & 15;

  v8f acc[2][8];
#pragma unroll
  for (int i = 0; i < 2; ++i)
#pragma unroll
    for (int j = 0; j < 8; ++j) acc[i][j] = zero_v8f();

  const bf16_t* arow0 = x + (size_t)(m0 + lm) * DD + half * 8;
  const bf16_t* arow1 = arow0 + (size_t)16 * DD;
  const bf16_t* brow[8];
#pragma unroll
  for (int j = 0; j < 8; ++j)
    brow[j] = W + (size_t)(n0 + j * 16 + lm) * DD + half * 8;

  for (int k0 = 0; k0 < DD; k0 += 32) {
    // Unconditional speculative prefetch of next A stripe (OOB is dropped).
    __builtin_prefetch(arow0 + k0 + 32, 0, 1);
    __builtin_prefetch(arow1 + k0 + 32, 0, 1);
    v16bf af0 = frag_from_bf16(arow0 + k0);
    v16bf af1 = frag_from_bf16(arow1 + k0);
#pragma unroll
    for (int j = 0; j < 8; ++j) {
      v16bf bfr = frag_from_bf16(brow[j] + k0);
      acc[0][j] = wmma_bf16(af0, bfr, acc[0][j]);
      acc[1][j] = wmma_bf16(af1, bfr, acc[1][j]);
    }
  }

  // C/D layout: lane col n = lm, rows m = r + 8*half (r = VGPR index).
#pragma unroll
  for (int j = 0; j < 8; ++j) {
    const int n = n0 + j * 16 + lm;
    const float bj = bias[n];
#pragma unroll
    for (int i = 0; i < 2; ++i) {
#pragma unroll
      for (int r = 0; r < 8; ++r) {
        const float vv = acc[i][j][r] + bj;
        const int m = m0 + i * 16 + r + 8 * half;
        if (MODE == 2) {
          out_f[(size_t)m * DD + n] = vv;
        } else {
          const int bb = m >> 11, s = m & (SS - 1);
          const int h = n >> 6,  dk = n & (DKK - 1);
          if (MODE == 0)
            out_bf[(((size_t)(bb * HH + h)) * SS + s) * DKK + dk] = (__bf16)vv;
          else
            out_bf[(((size_t)(bb * HH + h)) * DKK + dk) * SS + s] = (__bf16)vv;
        }
      }
    }
  }
}

// Flash attention, fully transposed: S^T = K@Q^T + maskbias, O^T = V^T@P^T.
// One wave per (b,h, 32-query block = two 16-query tiles; K/V frags shared).
// Lane owns query column lm; 32 keys/step. Mask enters as the WMMA
// C-initializer (additive -1e9 bias), so no per-element compare/select.
__global__ __launch_bounds__(256)
void attn_fwd(const bf16_t* __restrict__ Qh, const bf16_t* __restrict__ Kh,
              const bf16_t* __restrict__ Vt, const float* __restrict__ mbias,
              bf16_t* __restrict__ X) {
  const int lane = threadIdx.x & 31;
  const int wave = threadIdx.x >> 5;
  const int wid = blockIdx.x * 8 + wave;      // 0 .. B*H*(S/32)-1 = 2047
  const int qt = wid & (SS / 32 - 1);         // 64 query blocks
  const int bh = wid >> 6;                    // 0..31
  const int q0 = qt * 32;
  const int half = lane >> 4, lm = lane & 15;
  const int b = bh >> 4, h = bh & 15;

  const bf16_t* Qb = Qh + (size_t)bh * SS * DKK;
  const bf16_t* Kb = Kh + (size_t)bh * SS * DKK;
  const bf16_t* Vb = Vt + (size_t)bh * DKK * SS;

  // Q^T as B-fragments (lane = query col lm), pre-scaled by 1/sqrt(64)
  // (exact power-of-two scale in bf16).
  v16bf qf[2][2];
#pragma unroll
  for (int u = 0; u < 2; ++u) {
    const bf16_t* qrow = Qb + (size_t)(q0 + u * 16 + lm) * DKK + half * 8;
    qf[u][0] = frag_from_bf16(qrow);
    qf[u][1] = frag_from_bf16(qrow + 32);
#pragma unroll
    for (int i = 0; i < 16; ++i) {
      qf[u][0][i] = (__bf16)((float)qf[u][0][i] * 0.125f);
      qf[u][1][i] = (__bf16)((float)qf[u][1][i] * 0.125f);
    }
  }

  v8f o[2][4];      // O^T tiles: rows = dk (j*16 + r + 8*half), col = query lm
#pragma unroll
  for (int u = 0; u < 2; ++u)
#pragma unroll
    for (int j = 0; j < 4; ++j) o[u][j] = zero_v8f();
  float mcur[2] = {-3.0e38f, -3.0e38f};
  float lcur[2] = {0.0f, 0.0f};

  const float* bbase[2];
#pragma unroll
  for (int u = 0; u < 2; ++u)
    bbase[u] = mbias + (size_t)(q0 + u * 16 + lm) * SS + 8 * half;

  for (int kt = 0; kt < SS; kt += 32) {
    // K fragments for 32 keys (shared by both query tiles).
    const bf16_t* kr0 = Kb + (size_t)(kt + lm) * DKK + half * 8;
    const bf16_t* kr1 = kr0 + (size_t)16 * DKK;
    __builtin_prefetch(kr0 + 32 * DKK, 0, 1);         // next key block
    __builtin_prefetch(kr1 + 32 * DKK, 0, 1);
    const v16bf kf0a = frag_from_bf16(kr0);
    const v16bf kf0b = frag_from_bf16(kr0 + 32);
    const v16bf kf1a = frag_from_bf16(kr1);
    const v16bf kf1b = frag_from_bf16(kr1 + 32);

    v16bf pf[2];
#pragma unroll
    for (int u = 0; u < 2; ++u) {
      // Mask bias as C-init: per lane contiguous in key (aligned v8f loads).
      const float* bp = bbase[u] + kt;
      const v8f c0 = *(const v8f*)(bp);         // tile0 rows r=0..7
      const v8f c1 = *(const v8f*)(bp + 16);    // tile1 rows r=0..7

      v8f s0 = wmma_bf16(kf0a, qf[u][0], c0);
      s0 = wmma_bf16(kf0b, qf[u][1], s0);
      v8f s1 = wmma_bf16(kf1a, qf[u][0], c1);
      s1 = wmma_bf16(kf1b, qf[u][1], s1);

      float cmax = fmaxf(vmax8(s0), vmax8(s1));
      cmax = fmaxf(cmax, __shfl_xor(cmax, 16));       // other half's 16 keys
      const float mnew  = fmaxf(mcur[u], cmax);
      const float alpha = __expf(mcur[u] - mnew);
      mcur[u] = mnew;

      float e0[8], e1[8];
#pragma unroll
      for (int r = 0; r < 8; ++r) {
        e0[r] = __expf(s0[r] - mnew);
        e1[r] = __expf(s1[r] - mnew);
        pf[u][r]     = (__bf16)e0[r];                 // B-fragment of P^T
        pf[u][8 + r] = (__bf16)e1[r];
      }
      // Depth-4 tree sum of the 16 exps.
      const float t0 = (e0[0] + e0[1]) + (e0[2] + e0[3]);
      const float t1 = (e0[4] + e0[5]) + (e0[6] + e0[7]);
      const float t2 = (e1[0] + e1[1]) + (e1[2] + e1[3]);
      const float t3 = (e1[4] + e1[5]) + (e1[6] + e1[7]);
      float sum = (t0 + t1) + (t2 + t3);
      sum += __shfl_xor(sum, 16);
      lcur[u] = lcur[u] * alpha + sum;
#pragma unroll
      for (int j = 0; j < 4; ++j)
#pragma unroll
        for (int r = 0; r < 8; ++r) o[u][j][r] *= alpha;  // per-query = per-lane
    }

    // O^T += V^T @ P^T; V fragments shared by both query tiles.
#pragma unroll
    for (int j = 0; j < 4; ++j) {
      const bf16_t* vr = Vb + (size_t)(j * 16 + lm) * SS + kt + half * 8;
      __builtin_prefetch(vr + 32, 0, 1);
      const v16bf vfr = frag_from_bf16(vr);
      o[0][j] = wmma_bf16(vfr, pf[0], o[0][j]);
      o[1][j] = wmma_bf16(vfr, pf[1], o[1][j]);
    }
  }

  // Epilogue: divide by l, store bf16 X[b,s,d]; rows of O^T are contiguous dk.
#pragma unroll
  for (int u = 0; u < 2; ++u) {
    const float inv = lcur[u] > 0.0f ? 1.0f / lcur[u] : 0.0f;
    bf16_t* xrow = X + (size_t)(b * SS + q0 + u * 16 + lm) * DD
                     + h * DKK + 8 * half;
#pragma unroll
    for (int j = 0; j < 4; ++j) {
      v8bf ov;
#pragma unroll
      for (int r = 0; r < 8; ++r) ov[r] = (__bf16)(o[u][j][r] * inv);
      *(v8bf*)(xrow + j * 16) = ov;
    }
  }
}

extern "C" void kernel_launch(void* const* d_in, const int* in_sizes, int n_in,
                              void* d_out, int out_size, void* d_ws, size_t ws_size,
                              hipStream_t stream) {
  (void)in_sizes; (void)n_in; (void)out_size; (void)ws_size;
  const float* q   = (const float*)d_in[0];
  const float* k   = (const float*)d_in[1];
  const float* v   = (const float*)d_in[2];
  const int*  mask = (const int*)  d_in[3];
  const float* Wq  = (const float*)d_in[4];
  const float* bq  = (const float*)d_in[5];
  const float* Wk  = (const float*)d_in[6];
  const float* bk  = (const float*)d_in[7];
  const float* Wv  = (const float*)d_in[8];
  const float* bv  = (const float*)d_in[9];
  const float* Wo  = (const float*)d_in[10];
  const float* bo  = (const float*)d_in[11];
  float* out = (float*)d_out;

  // Workspace: qb/kb/vb 3x8MB, weights 4x2MB, Qh/Kh/Vt 3x8MB, X 8MB (bf16)
  //            + mask bias f32 16MB  => 80MB total.
  const size_t NA = (size_t)MM * DD;     // 4,194,304
  const size_t NW = (size_t)DD * DD;     // 1,048,576
  bf16_t* qb = (bf16_t*)d_ws;
  bf16_t* kb = qb + NA;
  bf16_t* vb = kb + NA;
  bf16_t* wq = vb + NA;
  bf16_t* wk = wq + NW;
  bf16_t* wv = wk + NW;
  bf16_t* wo = wv + NW;
  bf16_t* Qh = wo + NW;
  bf16_t* Kh = Qh + NA;
  bf16_t* Vt = Kh + NA;
  bf16_t* Xb = Vt + NA;
  float*  MBf = (float*)(Xb + NA);

  const int na4 = (int)(NA / 4), nw4 = (int)(NW / 4);
  const int nm4 = (SS * SS) / 4;
  cvt_bf16<<<na4 / 256, 256, 0, stream>>>(q,  qb, na4);
  cvt_bf16<<<na4 / 256, 256, 0, stream>>>(k,  kb, na4);
  cvt_bf16<<<na4 / 256, 256, 0, stream>>>(v,  vb, na4);
  cvt_bf16<<<nw4 / 256, 256, 0, stream>>>(Wq, wq, nw4);
  cvt_bf16<<<nw4 / 256, 256, 0, stream>>>(Wk, wk, nw4);
  cvt_bf16<<<nw4 / 256, 256, 0, stream>>>(Wv, wv, nw4);
  cvt_bf16<<<nw4 / 256, 256, 0, stream>>>(Wo, wo, nw4);
  mask_bias<<<nm4 / 256, 256, 0, stream>>>(mask, MBf, nm4);

  const int gemm_blocks = (MM / 32) * (DD / 128) / 8;   // 128 blocks x 8 waves
  proj_gemm<0><<<gemm_blocks, 256, 0, stream>>>(qb, wq, bq, Qh, nullptr);
  proj_gemm<0><<<gemm_blocks, 256, 0, stream>>>(kb, wk, bk, Kh, nullptr);
  proj_gemm<1><<<gemm_blocks, 256, 0, stream>>>(vb, wv, bv, Vt, nullptr);
  attn_fwd<<<(BB * HH * (SS / 32)) / 8, 256, 0, stream>>>(Qh, Kh, Vt, MBf, Xb);
  proj_gemm<2><<<gemm_blocks, 256, 0, stream>>>(Xb, wo, bo, nullptr, out);
}
